// SwinStack_38826504356007
// MI455X (gfx1250) — matrix operand order repaired and checked by reading, not compiled, not run on hardware
//
#include <hip/hip_runtime.h>

// ---------------------------------------------------------------------------
// Swin 3D transformer stack (2 layers) for gfx1250 (MI455X).
// All GEMMs run on v_wmma_f32_16x16x32_f16 (f16 inputs, f32 accumulate).
// ---------------------------------------------------------------------------

typedef _Float16 h16v __attribute__((ext_vector_type(16)));
typedef _Float16 h8v  __attribute__((ext_vector_type(8)));
typedef float    f8v  __attribute__((ext_vector_type(8)));

#define DIM      96
#define NHEAD    3
#define HD       32
#define NTOK     343     // 7*7*7 tokens per window
#define NPAD     352     // padded to 22 * 16
#define NWIN     128     // B(2) * 4*4*4 windows
#define M_TOT    43904   // NWIN * NTOK == 2*28^3 tokens
#define SHIFT3   3
#define NBIAS    2197    // 13^3

// -------------------------- wave reductions (wave32) -----------------------
__device__ __forceinline__ float wave_max(float v) {
#pragma unroll
  for (int m = 16; m >= 1; m >>= 1) v = fmaxf(v, __shfl_xor(v, m, 32));
  return v;
}
__device__ __forceinline__ float wave_sum(float v) {
#pragma unroll
  for (int m = 16; m >= 1; m >>= 1) v += __shfl_xor(v, m, 32);
  return v;
}

// -------------------------- f32 -> f16 weight convert ----------------------
__global__ void cvt_kernel(const float* __restrict__ in, _Float16* __restrict__ out, int n) {
  int i = blockIdx.x * 256 + threadIdx.x;
  if (i < n) out[i] = (_Float16)in[i];
}

// ---------------------------------------------------------------------------
// Build window-order <-> voxel permutation tables.
// map[T] = natural token index of the voxel that window-ordered token T
// touches: voxel = (window_origin + local + shift) mod 28 per dim.
// Same table serves LN1 gather (read source) and proj scatter (write dest).
// ---------------------------------------------------------------------------
__global__ void build_map_kernel(int* __restrict__ map0, int* __restrict__ map3) {
  int T = blockIdx.x * 256 + threadIdx.x;
  if (T >= M_TOT) return;
  int w = T / NTOK, l = T % NTOK;
  int b = w >> 6, wrem = w & 63;
  int jd = (wrem >> 4) * 7 + l / 49;
  int jh = ((wrem >> 2) & 3) * 7 + (l / 7) % 7;
  int jw = (wrem & 3) * 7 + l % 7;
  map0[T] = ((b * 28 + jd) * 28 + jh) * 28 + jw;
  int sd = (jd + SHIFT3) % 28, sh = (jh + SHIFT3) % 28, sw = (jw + SHIFT3) % 28;
  map3[T] = ((b * 28 + sd) * 28 + sh) * 28 + sw;
}

// ---------------------------------------------------------------------------
// LayerNorm, f32 -> f16. map != nullptr: gather source voxel from map[T]
// (fused shift-roll + window partition); else natural token order.
// One wave per token: 96 channels = 3 per lane.
// ---------------------------------------------------------------------------
__global__ void ln_kernel(const float* __restrict__ x, const float* __restrict__ g,
                          const float* __restrict__ bta, _Float16* __restrict__ out,
                          const int* __restrict__ map) {
  int wave = threadIdx.x >> 5, lane = threadIdx.x & 31;
  int T = blockIdx.x * 8 + wave;
  int src = map ? map[T] : T;
  const float* xp = x + (size_t)src * DIM;
  float v0 = xp[lane], v1 = xp[lane + 32], v2 = xp[lane + 64];
  float s  = wave_sum(v0 + v1 + v2);
  float ss = wave_sum(v0 * v0 + v1 * v1 + v2 * v2);
  float mean = s * (1.0f / 96.0f);
  float var  = ss * (1.0f / 96.0f) - mean * mean;
  float inv  = rsqrtf(var + 1e-5f);
  _Float16* op = out + (size_t)T * DIM;
  op[lane]      = (_Float16)((v0 - mean) * inv * g[lane]      + bta[lane]);
  op[lane + 32] = (_Float16)((v1 - mean) * inv * g[lane + 32] + bta[lane + 32]);
  op[lane + 64] = (_Float16)((v2 - mean) * inv * g[lane + 64] + bta[lane + 64]);
}

// ---------------------------------------------------------------------------
// Tiled WMMA GEMM: C[M,N] = A[M,K] @ W[N,K]^T + bias, fused epilogue.
//  MODE 0: store f16 (QKV)
//  MODE 1: exact GELU, store f16 (fc1)
//  MODE 2: scatter via map[] (window-reverse + roll), residual add into Xres
//  MODE 3: residual add into Xres, natural order (fc2)
// WG = 128 threads (4 waves); tile M=128, N=32, K-step 32.
// LDS row stride 40 halfs: 20-dword lane stride -> conflict-free b128 reads.
// ---------------------------------------------------------------------------
#define LSTR 40
template <int MODE>
__global__ __launch_bounds__(128)
void gemm_kernel(const _Float16* __restrict__ A, const _Float16* __restrict__ W,
                 const float* __restrict__ bias, _Float16* __restrict__ O16,
                 float* __restrict__ Xres, const int* __restrict__ map,
                 int K, int N) {
  __shared__ _Float16 As[128 * LSTR];
  __shared__ _Float16 Ws[32 * LSTR];
  const int tid  = threadIdx.x;
  const int wave = tid >> 5, lane = tid & 31;
  const int hl = (lane >> 4) & 1, l16 = lane & 15;
  const int m0 = blockIdx.x * 128, n0 = blockIdx.y * 32;

  f8v acc[2][2];
#pragma unroll
  for (int mi = 0; mi < 2; ++mi)
#pragma unroll
    for (int ni = 0; ni < 2; ++ni) {
      f8v z = {0.f, 0.f, 0.f, 0.f, 0.f, 0.f, 0.f, 0.f};
      acc[mi][ni] = z;
    }

  const int nk = K >> 5;
  for (int kc = 0; kc < nk; ++kc) {
    __syncthreads();
    // stage A tile: 128 rows x 32 halfs, one row per thread
    {
      const h8v* src = (const h8v*)(A + (size_t)(m0 + tid) * K + kc * 32);
      *(h8v*)(As + tid * LSTR)     = src[0];
      *(h8v*)(As + tid * LSTR + 8) = src[1];
    }
    // stage W tile: 32 rows x 32 halfs
    {
      int wr = tid >> 2, sub = tid & 3;
      h8v wv = *(const h8v*)(W + (size_t)(n0 + wr) * K + kc * 32 + sub * 8);
      *(h8v*)(Ws + wr * LSTR + sub * 8) = wv;
    }
    __syncthreads();

    h16v af[2], bf[2];
#pragma unroll
    for (int mi = 0; mi < 2; ++mi) {
      const _Float16* p = As + (wave * 32 + mi * 16 + l16) * LSTR;
      *(h8v*)&af[mi]       = *(const h8v*)(p + 8 * hl);
      *((h8v*)&af[mi] + 1) = *(const h8v*)(p + 16 + 8 * hl);
    }
#pragma unroll
    for (int ni = 0; ni < 2; ++ni) {
      const _Float16* p = Ws + (ni * 16 + l16) * LSTR;
      *(h8v*)&bf[ni]       = *(const h8v*)(p + 8 * hl);
      *((h8v*)&bf[ni] + 1) = *(const h8v*)(p + 16 + 8 * hl);
    }
#pragma unroll
    for (int mi = 0; mi < 2; ++mi)
#pragma unroll
      for (int ni = 0; ni < 2; ++ni)
        acc[mi][ni] = __builtin_amdgcn_wmma_f32_16x16x32_f16(
            false, af[mi], false, bf[ni], (short)0, acc[mi][ni], false, false);
  }

  // epilogue (branch-free per instantiation)
  const int mbase = m0 + wave * 32 + 8 * hl;
#pragma unroll
  for (int mi = 0; mi < 2; ++mi)
#pragma unroll
    for (int ni = 0; ni < 2; ++ni) {
      int nn = n0 + ni * 16 + l16;
      float bv = bias[nn];
#pragma unroll
      for (int e = 0; e < 8; ++e) {
        int m = mbase + mi * 16 + e;
        float v = acc[mi][ni][e] + bv;
        if (MODE == 0) {
          O16[(size_t)m * N + nn] = (_Float16)v;
        } else if (MODE == 1) {
          v = 0.5f * v * (1.0f + erff(v * 0.70710678118f));
          O16[(size_t)m * N + nn] = (_Float16)v;
        } else if (MODE == 2) {
          int dst = map[m];
          Xres[(size_t)dst * DIM + nn] += v;
        } else {
          Xres[(size_t)m * DIM + nn] += v;
        }
      }
    }
}

// ---------------------------------------------------------------------------
// Windowed attention, one WG per (window, head). 128 threads / 4 waves.
// q/k/vT held in LDS as f16 (zero-padded 343->352); per-head rel-bias table
// (2197 f32) staged in LDS; each wave owns 16-query row blocks: QK^T via
// WMMA, branchless rel-bias + shift-mask epilogue from packed per-token
// metadata (region<<16 | d<<8 | h<<4 | w), softmax with wave reductions,
// PV via WMMA (K=352 in 11 steps of 32). ~226 KB LDS per WGP.
// ---------------------------------------------------------------------------
#define QSTR 40    // q/k row stride (halfs)
#define VSTR 360   // vT row stride (halfs)
#define SSTR 352   // scores row stride (floats)
#define PSTR 360   // probs row stride (halfs)
#define ATTN_SMEM (NPAD*QSTR*2 /*qs*/ + NPAD*QSTR*2 /*ks*/ + 32*VSTR*2 /*vT*/ \
                 + 4*16*SSTR*4 /*scores*/ + 4*16*PSTR*2 /*probs*/ \
                 + NPAD*4 /*meta*/ + NBIAS*4 /*bias*/)

__global__ __launch_bounds__(128)
void attn_kernel(const _Float16* __restrict__ qkv, const float* __restrict__ rel,
                 _Float16* __restrict__ outp, int use_mask) {
  extern __shared__ char smem[];
  _Float16* qs = (_Float16*)smem;                    // [352][40]
  _Float16* ks = qs + NPAD * QSTR;                   // [352][40]
  _Float16* vT = ks + NPAD * QSTR;                   // [32][360]
  float*    sc = (float*)(vT + 32 * VSTR);           // 4 x [16][352]
  _Float16* pr = (_Float16*)(sc + 4 * 16 * SSTR);    // 4 x [16][360]
  int*    meta = (int*)(pr + 4 * 16 * PSTR);         // [352] packed coords+region
  float* relL  = (float*)(meta + NPAD);              // [2197] this head's bias

  const int wid = blockIdx.x;
  const int w = wid / NHEAD, head = wid % NHEAD;
  const int tid = threadIdx.x, wave = tid >> 5, lane = tid & 31;
  const int hl = (lane >> 4) & 1, l16 = lane & 15;
  const int wrem = w & 63;
  const int od = (wrem >> 4) * 7, oh = ((wrem >> 2) & 3) * 7, ow = (wrem & 3) * 7;
  const float scale = 0.17677669529663689f; // 32^-0.5
  const float* relh = rel + head;

  // ---- stage this head's relative-position bias table into LDS ----
  for (int idx = tid; idx < NBIAS; idx += 128) relL[idx] = relh[idx * NHEAD];

  // ---- stage q (scaled), k, vT into LDS as f16; zero rows/cols >= 343 ----
  for (int t = tid; t < NPAD; t += 128) {
    if (t < NTOK) {
      const _Float16* base = qkv + (size_t)(w * NTOK + t) * 288 + head * HD;
      _Float16 qv[32];
      *(h8v*)(qv)      = *(const h8v*)(base);
      *(h8v*)(qv + 8)  = *(const h8v*)(base + 8);
      *(h8v*)(qv + 16) = *(const h8v*)(base + 16);
      *(h8v*)(qv + 24) = *(const h8v*)(base + 24);
#pragma unroll
      for (int d = 0; d < 32; ++d) qs[t * QSTR + d] = (_Float16)((float)qv[d] * scale);
      const _Float16* kb = base + DIM;     // k at +96
#pragma unroll
      for (int j = 0; j < 4; ++j)
        *(h8v*)(ks + t * QSTR + j * 8) = *(const h8v*)(kb + j * 8);
      const _Float16* vb = base + 2 * DIM; // v at +192
      _Float16 vv[32];
      *(h8v*)(vv)      = *(const h8v*)(vb);
      *(h8v*)(vv + 8)  = *(const h8v*)(vb + 8);
      *(h8v*)(vv + 16) = *(const h8v*)(vb + 16);
      *(h8v*)(vv + 24) = *(const h8v*)(vb + 24);
#pragma unroll
      for (int d = 0; d < 32; ++d) vT[d * VSTR + t] = vv[d];
    } else {
#pragma unroll
      for (int d = 0; d < 32; ++d) {
        qs[t * QSTR + d] = (_Float16)0.0f;
        ks[t * QSTR + d] = (_Float16)0.0f;
        vT[d * VSTR + t] = (_Float16)0.0f;
      }
    }
    // packed per-token metadata: local coords + shift-window region id
    // (zero for padded tokens -> bias index stays in range; masked anyway)
    int mt = 0;
    if (t < NTOK) {
      int ld = t / 49, lh = (t / 7) % 7, lw = t % 7;
      int cd = od + ld, ch = oh + lh, cw = ow + lw;
      int rd = cd < 21 ? 0 : (cd < 25 ? 1 : 2);
      int rh = ch < 21 ? 0 : (ch < 25 ? 1 : 2);
      int rw = cw < 21 ? 0 : (cw < 25 ? 1 : 2);
      mt = ((rd * 9 + rh * 3 + rw) << 16) | (ld << 8) | (lh << 4) | lw;
    }
    meta[t] = mt;
  }
  __syncthreads();

  float*    msc = sc + wave * 16 * SSTR;
  _Float16* mpr = pr + wave * 16 * PSTR;

  for (int rb = wave; rb < 22; rb += 4) {
    // A fragment: 16 query rows (rb*16 .. +15), K = 32 (head dim)
    h16v aq;
    {
      const _Float16* p = qs + (rb * 16 + l16) * QSTR;
      *(h8v*)&aq       = *(const h8v*)(p + 8 * hl);
      *((h8v*)&aq + 1) = *(const h8v*)(p + 16 + 8 * hl);
    }
    // hoist this lane's 8 query-token metadata words out of the key loop
    int metai[8];
#pragma unroll
    for (int e = 0; e < 8; ++e) metai[e] = meta[rb * 16 + e + 8 * hl];

    // ---- scores = q @ k^T, + rel bias + mask (branchless), per-wave LDS ----
    for (int kt = 0; kt < 22; ++kt) {
      h16v bk;
      {
        const _Float16* p = ks + (kt * 16 + l16) * QSTR;
        *(h8v*)&bk       = *(const h8v*)(p + 8 * hl);
        *((h8v*)&bk + 1) = *(const h8v*)(p + 16 + 8 * hl);
      }
      f8v a = {0.f, 0.f, 0.f, 0.f, 0.f, 0.f, 0.f, 0.f};
      a = __builtin_amdgcn_wmma_f32_16x16x32_f16(false, aq, false, bk, (short)0, a, false, false);
      const int j = kt * 16 + l16;
      const bool jvalid = (j < NTOK);
      const int mj = meta[j];
      const int dj = (mj >> 8) & 15, hj = (mj >> 4) & 15, wj = mj & 15;
#pragma unroll
      for (int e = 0; e < 8; ++e) {
        int mi_ = metai[e];
        int di = ((mi_ >> 8) & 15) - dj + 6;
        int hi = ((mi_ >> 4) & 15) - hj + 6;
        int wi = (mi_ & 15) - wj + 6;
        float s = a[e] + relL[(di * 13 + hi) * 13 + wi];     // always in range
        if (use_mask && (((mi_ ^ mj) >> 16) != 0)) s -= 100.0f;  // v_cndmask
        if (!jvalid) s = -1e30f;                                  // v_cndmask
        msc[(e + 8 * hl) * SSTR + j] = s;
      }
    }
    // ---- softmax over 352 (padded cols contribute exp(-inf)=0) ----
    for (int r = 0; r < 16; ++r) {
      float mx = -1e30f;
      for (int k = lane; k < NPAD; k += 32) mx = fmaxf(mx, msc[r * SSTR + k]);
      mx = wave_max(mx);
      float sum = 0.0f;
      for (int k = lane; k < NPAD; k += 32) {
        float ev = __expf(msc[r * SSTR + k] - mx);
        msc[r * SSTR + k] = ev;
        sum += ev;
      }
      sum = wave_sum(sum);
      float inv = 1.0f / sum;
      for (int k = lane; k < NPAD; k += 32)
        mpr[r * PSTR + k] = (_Float16)(msc[r * SSTR + k] * inv);
    }
    // ---- out = probs @ v : K = 352 in 11 WMMA steps, N = 32 ----
    f8v o0 = {0.f, 0.f, 0.f, 0.f, 0.f, 0.f, 0.f, 0.f};
    f8v o1 = o0;
    for (int kc = 0; kc < 11; ++kc) {
      h16v ap;
      {
        const _Float16* p = mpr + l16 * PSTR + kc * 32;
        *(h8v*)&ap       = *(const h8v*)(p + 8 * hl);
        *((h8v*)&ap + 1) = *(const h8v*)(p + 16 + 8 * hl);
      }
      h16v bv;
      {
        const _Float16* p = vT + (l16)*VSTR + kc * 32;
        *(h8v*)&bv       = *(const h8v*)(p + 8 * hl);
        *((h8v*)&bv + 1) = *(const h8v*)(p + 16 + 8 * hl);
      }
      o0 = __builtin_amdgcn_wmma_f32_16x16x32_f16(false, ap, false, bv, (short)0, o0, false, false);
      {
        const _Float16* p = vT + (16 + l16) * VSTR + kc * 32;
        *(h8v*)&bv       = *(const h8v*)(p + 8 * hl);
        *((h8v*)&bv + 1) = *(const h8v*)(p + 16 + 8 * hl);
      }
      o1 = __builtin_amdgcn_wmma_f32_16x16x32_f16(false, ap, false, bv, (short)0, o1, false, false);
    }
#pragma unroll
    for (int e = 0; e < 8; ++e) {
      int i = rb * 16 + e + 8 * hl;
      if (i < NTOK) {
        size_t rowo = (size_t)(w * NTOK + i) * DIM + head * HD;
        outp[rowo + l16]      = (_Float16)o0[e];
        outp[rowo + 16 + l16] = (_Float16)o1[e];
      }
    }
  }
}

// ---------------------------------------------------------------------------
extern "C" void kernel_launch(void* const* d_in, const int* in_sizes, int n_in,
                              void* d_out, int out_size, void* d_ws, size_t ws_size,
                              hipStream_t stream) {
  const float* x_in  = (const float*)d_in[0];
  const float* n1g   = (const float*)d_in[1];
  const float* n1b   = (const float*)d_in[2];
  const float* qkvw  = (const float*)d_in[3];
  const float* qkvb  = (const float*)d_in[4];
  const float* relb  = (const float*)d_in[5];
  const float* projw = (const float*)d_in[6];
  const float* projb = (const float*)d_in[7];
  const float* n2g   = (const float*)d_in[8];
  const float* n2b   = (const float*)d_in[9];
  const float* fc1w  = (const float*)d_in[10];
  const float* fc1b  = (const float*)d_in[11];
  const float* fc2w  = (const float*)d_in[12];
  const float* fc2b  = (const float*)d_in[13];
  float* x = (float*)d_out;   // residual stream lives in the output buffer

  char* ws = (char*)d_ws;
  size_t off = 0;
  auto carve = [&](size_t bytes) -> char* {
    char* p = ws + off;
    off = (off + bytes + 255) & ~(size_t)255;
    return p;
  };
  _Float16* h16f  = (_Float16*)carve((size_t)M_TOT * DIM * 2);   // LN output (f16)
  _Float16* qkv16 = (_Float16*)carve((size_t)M_TOT * 288 * 2);   // QKV (f16)
  _Float16* ao16  = (_Float16*)carve((size_t)M_TOT * DIM * 2);   // attn out (f16)
  _Float16* f116  = (_Float16*)carve((size_t)M_TOT * 384 * 2);   // fc1 out (f16)
  int*      map0  = (int*)carve((size_t)M_TOT * 4);              // perm, shift=0
  int*      map3  = (int*)carve((size_t)M_TOT * 4);              // perm, shift=3
  _Float16* wq16  = (_Float16*)carve(288 * 96 * 2);
  _Float16* wp16  = (_Float16*)carve(96 * 96 * 2);
  _Float16* w116  = (_Float16*)carve(384 * 96 * 2);
  _Float16* w216  = (_Float16*)carve(96 * 384 * 2);

  hipMemcpyAsync(x, x_in, (size_t)M_TOT * DIM * sizeof(float),
                 hipMemcpyDeviceToDevice, stream);
  build_map_kernel<<<(M_TOT + 255) / 256, 256, 0, stream>>>(map0, map3);

  for (int L = 0; L < 2; ++L) {
    const int use_mask = (L & 1);
    const int* map = use_mask ? map3 : map0;

    cvt_kernel<<<(288 * 96 + 255) / 256, 256, 0, stream>>>(qkvw + (size_t)L * 288 * 96, wq16, 288 * 96);
    cvt_kernel<<<(96 * 96 + 255) / 256, 256, 0, stream>>>(projw + (size_t)L * 96 * 96, wp16, 96 * 96);
    cvt_kernel<<<(384 * 96 + 255) / 256, 256, 0, stream>>>(fc1w + (size_t)L * 384 * 96, w116, 384 * 96);
    cvt_kernel<<<(96 * 384 + 255) / 256, 256, 0, stream>>>(fc2w + (size_t)L * 96 * 384, w216, 96 * 384);

    // LN1 + shift + window partition (gather through map)
    ln_kernel<<<M_TOT / 8, 256, 0, stream>>>(x, n1g + L * DIM, n1b + L * DIM, h16f, map);
    // QKV: [43904,96] @ [96,288]
    gemm_kernel<0><<<dim3(M_TOT / 128, 288 / 32), 128, 0, stream>>>(
        h16f, wq16, qkvb + L * 288, qkv16, nullptr, nullptr, 96, 288);
    // windowed attention
    attn_kernel<<<NWIN * NHEAD, 128, ATTN_SMEM, stream>>>(
        qkv16, relb + (size_t)L * NBIAS * NHEAD, ao16, use_mask);
    // proj + window-reverse + roll-back scatter + residual add into x
    gemm_kernel<2><<<dim3(M_TOT / 128, 96 / 32), 128, 0, stream>>>(
        ao16, wp16, projb + L * DIM, nullptr, x, map, 96, 96);
    // LN2 (natural order)
    ln_kernel<<<M_TOT / 8, 256, 0, stream>>>(x, n2g + L * DIM, n2b + L * DIM, h16f, nullptr);
    // fc1 + exact GELU
    gemm_kernel<1><<<dim3(M_TOT / 128, 384 / 32), 128, 0, stream>>>(
        h16f, w116, fc1b + L * 384, f116, nullptr, nullptr, 96, 384);
    // fc2 + residual add into x
    gemm_kernel<3><<<dim3(M_TOT / 128, 96 / 32), 128, 0, stream>>>(
        f116, w216, fc2b + L * DIM, nullptr, x, nullptr, 384, 96);
  }
}
